// MSAColumnGlobalAttention_39118562132440
// MI455X (gfx1250) — matrix operand order repaired and checked by reading, not compiled, not run on hardware
//
#include <hip/hip_runtime.h>
#include <hip/hip_bf16.h>
#include <math.h>

typedef _Float16 f16;
typedef __attribute__((ext_vector_type(16))) _Float16 v16h;
typedef __attribute__((ext_vector_type(8)))  _Float16 v8h;
typedef __attribute__((ext_vector_type(8)))  float    v8f;

#define NSEQ 2048
#define NRES 512
#define CIN  64
#define HC   64
#define BIGNEG  (-1e30f)
#define MASKINF 1000000000.0f

__device__ __forceinline__ float wave_sum32(float v) {
#pragma unroll
  for (int off = 16; off >= 1; off >>= 1) v += __shfl_xor(v, off, 32);
  return v;
}

// A-fragment (16x32 f16, M x K) gathered from a row-major 16x64 f16 LDS tile.
// Lane L holds row M=L&15; halves 0..7 = K base+0..7(+8 if L>=16), 8..15 = +16.
__device__ __forceinline__ v16h frag_a(const f16* xtile, int lane, int kc) {
  const int row = lane & 15;
  const f16* p = xtile + row * 64 + kc * 32 + 8 * (lane >> 4);
  v8h lo = *(const v8h*)p;
  v8h hi = *(const v8h*)(p + 16);
  return __builtin_shufflevector(lo, hi, 0,1,2,3,4,5,6,7,8,9,10,11,12,13,14,15);
}

// B-fragment (32x16 f16, K x N) from row-major [64][ncols] f16 LDS weights.
// Lane L holds col N = nc*16 + (L&15); half j = K = kc*32 + 8*(L>=16) + j (+8 if j>=8).
__device__ __forceinline__ v16h frag_b(const f16* w, int ncols, int lane, int kc, int nc) {
  const int col = nc * 16 + (lane & 15);
  const int kbase = kc * 32 + 8 * (lane >> 4);
  v16h b;
#pragma unroll
  for (int j = 0; j < 16; ++j) b[j] = w[(kbase + j + (j >= 8 ? 8 : 0)) * ncols + col];
  return b;
}

// ---------------------------------------------------------------------------
// Kernel 1: per residue r. One pass over m: LN -> 256KB LDS cache (CDNA5 WGP
// LDS). Global query from masked mean, k/v via WMMA from the LDS cache,
// per-head online softmax -> o[r]. Mask column staged via async-to-LDS.
// grid = NRES blocks, 256 threads (8 waves), wave w owns rows n = (w+8t)*16..+15
// ---------------------------------------------------------------------------
__global__ void __launch_bounds__(256) msa_colglob_qkvo_kernel(
    const float* __restrict__ m, const float* __restrict__ mask,
    const float* __restrict__ ln_w, const float* __restrict__ ln_b,
    const float* __restrict__ w_q, const float* __restrict__ w_k,
    const float* __restrict__ w_v, float* __restrict__ o_ws) {
  extern __shared__ __align__(16) char dyn_smem[];  // NSEQ x 64 f16 = 256 KB
  f16* xall = (f16*)dyn_smem;

  __shared__ f16   sh_wkv[64 * 16];       // [w_k | w_v] as 64x16 f16
  __shared__ float sh_mask[NSEQ];         // mask column (async staged)
  __shared__ float sh_kvt[8][16 * 16];    // per-wave k|v result tile (f32)
  __shared__ float sh_red[8][64];
  __shared__ float sh_cnt[8];
  __shared__ float sh_q[64];
  __shared__ float sh_qout[64];
  __shared__ float sh_m[64], sh_s[64];
  __shared__ float sh_oacc[8][64];

  const int r    = blockIdx.x;
  const int tid  = threadIdx.x;
  const int wave = tid >> 5;
  const int lane = tid & 31;

  // Async-stage the strided mask column into LDS (per-lane gather, ASYNCcnt).
  for (int idx = tid; idx < NSEQ; idx += 256) {
    const unsigned long long gp =
        (unsigned long long)(uintptr_t)(mask + (size_t)idx * NRES + r);
    const unsigned ldsoff = (unsigned)(uintptr_t)(&sh_mask[idx]);
    asm volatile("global_load_async_to_lds_b32 %0, %1, off"
                 :: "v"(ldsoff), "v"(gp) : "memory");
  }

  // stage [w_k | w_v] -> 64x16 f16
  for (int idx = tid; idx < 512; idx += 256) {
    const int row = idx >> 3, c = idx & 7;
    sh_wkv[row * 16 + c]     = (f16)w_k[idx];
    sh_wkv[row * 16 + 8 + c] = (f16)w_v[idx];
  }

  const float lw0 = ln_w[2 * lane], lw1 = ln_w[2 * lane + 1];
  const float lb0 = ln_b[2 * lane], lb1 = ln_b[2 * lane + 1];

  asm volatile("s_wait_asynccnt 0x0" ::: "memory");
  __syncthreads();

  // ---- Pass A: single global read of m; LN -> LDS cache + masked mean ----
  float acc0 = 0.f, acc1 = 0.f, cnt = 0.f;
  for (int t = 0; t < 16; ++t) {
    const int n0 = (wave + 8 * t) * 16;
#pragma unroll 4
    for (int j = 0; j < 16; ++j) {
      const int n = n0 + j;
      const float* rowp = m + ((size_t)n * NRES + r) * CIN;
      if (j + 1 < 16)
        __builtin_prefetch(m + ((size_t)(n + 1) * NRES + r) * CIN + 2 * lane, 0, 1);
      float2 xv = *(const float2*)(rowp + 2 * lane);
      const float s   = wave_sum32(xv.x + xv.y);
      const float s2  = wave_sum32(xv.x * xv.x + xv.y * xv.y);
      const float mu  = s * (1.f / 64.f);
      const float rst = rsqrtf(s2 * (1.f / 64.f) - mu * mu + 1e-5f);
      const float x0  = (xv.x - mu) * rst * lw0 + lb0;
      const float x1  = (xv.y - mu) * rst * lw1 + lb1;
      xall[(size_t)n * 64 + 2 * lane]     = (f16)x0;
      xall[(size_t)n * 64 + 2 * lane + 1] = (f16)x1;
      const float mk = sh_mask[n];
      acc0 += mk * x0; acc1 += mk * x1; cnt += mk;
    }
  }
  sh_red[wave][2 * lane]     = acc0;
  sh_red[wave][2 * lane + 1] = acc1;
  if (lane == 0) sh_cnt[wave] = cnt;
  __syncthreads();
  if (tid < 64) {
    float qn = 0.f, den = 0.f;
#pragma unroll
    for (int w = 0; w < 8; ++w) { qn += sh_red[w][tid]; den += sh_cnt[w]; }
    sh_q[tid] = qn / (den + 1e-10f);
  }
  __syncthreads();
  if (tid < 64) {
    float qq = 0.f;
    for (int j = 0; j < 64; ++j) qq += sh_q[j] * w_q[j * 64 + tid];
    sh_qout[tid] = qq * 0.35355339059327373f;  // C^-0.5 = 1/sqrt(8)
  }
  __syncthreads();

  // ---- Pass B: k/v via WMMA from LDS cache + per-head online softmax ----
  const v16h bk0 = frag_b(sh_wkv, 16, lane, 0, 0);
  const v16h bk1 = frag_b(sh_wkv, 16, lane, 1, 0);

  const int h  = lane >> 2;          // 4 lanes per head
  const int c0 = (lane & 3) * 2;     // 2 channels per lane
  const float qh0 = sh_qout[h * 8 + c0];
  const float qh1 = sh_qout[h * 8 + c0 + 1];

  float mh = BIGNEG, ssum = 0.f, o0 = 0.f, o1 = 0.f;
  float* kvt = sh_kvt[wave];

  for (int t = 0; t < 16; ++t) {
    const int n0 = (wave + 8 * t) * 16;
    const f16* xt = xall + (size_t)n0 * 64;
    const v16h a0 = frag_a(xt, lane, 0);
    const v16h a1 = frag_a(xt, lane, 1);
    v8f acc = {};
    acc = __builtin_amdgcn_wmma_f32_16x16x32_f16(false, a0, false, bk0, (short)0, acc, false, false);
    acc = __builtin_amdgcn_wmma_f32_16x16x32_f16(false, a1, false, bk1, (short)0, acc, false, false);
#pragma unroll
    for (int v = 0; v < 8; ++v)
      kvt[(v + 8 * (lane >> 4)) * 16 + (lane & 15)] = acc[v];
    asm volatile("s_wait_dscnt 0x0" ::: "memory");
#pragma unroll 4
    for (int j = 0; j < 16; ++j) {
      const int n = n0 + j;
      const float mk = sh_mask[n];
      float p = qh0 * kvt[j * 16 + c0] + qh1 * kvt[j * 16 + c0 + 1];
      p += __shfl_xor(p, 1, 32);
      p += __shfl_xor(p, 2, 32);  // full 8-dim q.k within the 4-lane head group
      const float l  = p + MASKINF * (mk - 1.f);
      const float nm = fmaxf(mh, l);
      const float sc = __expf(mh - nm);
      const float pe = __expf(l - nm);
      ssum = ssum * sc + pe;
      o0   = o0 * sc + pe * kvt[j * 16 + 8 + c0];
      o1   = o1 * sc + pe * kvt[j * 16 + 8 + c0 + 1];
      mh   = nm;
    }
  }
  if ((lane & 3) == 0) { sh_m[wave * 8 + h] = mh; sh_s[wave * 8 + h] = ssum; }
  sh_oacc[wave][h * 8 + c0]     = o0;
  sh_oacc[wave][h * 8 + c0 + 1] = o1;
  __syncthreads();
  if (tid < 64) {
    const int hh = tid >> 3;
    float M = BIGNEG;
#pragma unroll
    for (int w = 0; w < 8; ++w) M = fmaxf(M, sh_m[w * 8 + hh]);
    float S = 0.f, O = 0.f;
#pragma unroll
    for (int w = 0; w < 8; ++w) {
      const float e = __expf(sh_m[w * 8 + hh] - M);
      S += sh_s[w * 8 + hh] * e;
      O += sh_oacc[w][tid] * e;
    }
    o_ws[(size_t)r * HC + tid] = O / S;
  }
}

// ---------------------------------------------------------------------------
// Kernel 2: g = sigmoid(x@w_g+b_g); out = (o*g)@w_o + b_o  — all via WMMA
// grid = (NSEQ/128, NRES), 256 threads; each wave owns one 16-row tile.
// ---------------------------------------------------------------------------
__global__ void __launch_bounds__(256) msa_colglob_out_kernel(
    const float* __restrict__ m,
    const float* __restrict__ ln_w, const float* __restrict__ ln_b,
    const float* __restrict__ w_g, const float* __restrict__ b_g,
    const float* __restrict__ w_o, const float* __restrict__ b_o,
    const float* __restrict__ o_ws, float* __restrict__ out) {
  __shared__ f16   sh_wg[64 * 64];
  __shared__ f16   sh_wo[64 * 64];
  __shared__ f16   sh_x[8][16 * 64];
  __shared__ float sh_bg[64], sh_bo[64], sh_o[64];

  const int r    = blockIdx.y;
  const int tid  = threadIdx.x;
  const int wave = tid >> 5;
  const int lane = tid & 31;

  for (int idx = tid; idx < 4096; idx += 256) {
    sh_wg[idx] = (f16)w_g[idx];
    sh_wo[idx] = (f16)w_o[idx];
  }
  if (tid < 64) {
    sh_bg[tid] = b_g[tid];
    sh_bo[tid] = b_o[tid];
    sh_o[tid]  = o_ws[(size_t)r * HC + tid];
  }
  __syncthreads();

  const float lw0 = ln_w[2 * lane], lw1 = ln_w[2 * lane + 1];
  const float lb0 = ln_b[2 * lane], lb1 = ln_b[2 * lane + 1];
  const int n0 = blockIdx.x * 128 + wave * 16;
  f16* xt = sh_x[wave];

#pragma unroll 4
  for (int j = 0; j < 16; ++j) {
    const int n = n0 + j;
    const float* rowp = m + ((size_t)n * NRES + r) * CIN;
    if (j + 1 < 16)
      __builtin_prefetch(m + ((size_t)(n + 1) * NRES + r) * CIN + 2 * lane, 0, 1);
    float2 xv = *(const float2*)(rowp + 2 * lane);
    const float s   = wave_sum32(xv.x + xv.y);
    const float s2  = wave_sum32(xv.x * xv.x + xv.y * xv.y);
    const float mu  = s * (1.f / 64.f);
    const float rst = rsqrtf(s2 * (1.f / 64.f) - mu * mu + 1e-5f);
    xt[j * 64 + 2 * lane]     = (f16)((xv.x - mu) * rst * lw0 + lb0);
    xt[j * 64 + 2 * lane + 1] = (f16)((xv.y - mu) * rst * lw1 + lb1);
  }
  asm volatile("s_wait_dscnt 0x0" ::: "memory");
  const v16h a0 = frag_a(xt, lane, 0);
  const v16h a1 = frag_a(xt, lane, 1);

  // y = sigmoid(x@w_g + b_g) * o  -> restaged into xt (A-layout source)
#pragma unroll
  for (int nc = 0; nc < 4; ++nc) {
    const v16h b0 = frag_b(sh_wg, 64, lane, 0, nc);
    const v16h b1 = frag_b(sh_wg, 64, lane, 1, nc);
    v8f d = {};
    d = __builtin_amdgcn_wmma_f32_16x16x32_f16(false, a0, false, b0, (short)0, d, false, false);
    d = __builtin_amdgcn_wmma_f32_16x16x32_f16(false, a1, false, b1, (short)0, d, false, false);
#pragma unroll
    for (int v = 0; v < 8; ++v) {
      const int M = v + 8 * (lane >> 4);
      const int N = nc * 16 + (lane & 15);
      const float g = 1.f / (1.f + __expf(-(d[v] + sh_bg[N])));
      xt[M * 64 + N] = (f16)(g * sh_o[N]);
    }
  }
  asm volatile("s_wait_dscnt 0x0" ::: "memory");
  const v16h ya0 = frag_a(xt, lane, 0);
  const v16h ya1 = frag_a(xt, lane, 1);

#pragma unroll
  for (int nc = 0; nc < 4; ++nc) {
    const v16h b0 = frag_b(sh_wo, 64, lane, 0, nc);
    const v16h b1 = frag_b(sh_wo, 64, lane, 1, nc);
    v8f d = {};
    d = __builtin_amdgcn_wmma_f32_16x16x32_f16(false, ya0, false, b0, (short)0, d, false, false);
    d = __builtin_amdgcn_wmma_f32_16x16x32_f16(false, ya1, false, b1, (short)0, d, false, false);
#pragma unroll
    for (int v = 0; v < 8; ++v) {
      const int M = v + 8 * (lane >> 4);
      const int N = nc * 16 + (lane & 15);
      out[((size_t)(n0 + M) * NRES + r) * CIN + N] = d[v] + sh_bo[N];
    }
  }
}

extern "C" void kernel_launch(void* const* d_in, const int* in_sizes, int n_in,
                              void* d_out, int out_size, void* d_ws, size_t ws_size,
                              hipStream_t stream) {
  const float* m    = (const float*)d_in[0];
  const float* mask = (const float*)d_in[1];
  const float* ln_w = (const float*)d_in[2];
  const float* ln_b = (const float*)d_in[3];
  const float* w_q  = (const float*)d_in[4];
  const float* w_k  = (const float*)d_in[5];
  const float* w_v  = (const float*)d_in[6];
  const float* w_g  = (const float*)d_in[7];
  const float* b_g  = (const float*)d_in[8];
  const float* w_o  = (const float*)d_in[9];
  const float* b_o  = (const float*)d_in[10];
  float* o_ws = (float*)d_ws;          // NRES * HC floats (128 KB)
  float* out  = (float*)d_out;

  // Kernel 1 caches the whole LN'd column in 256 KB of CDNA5 WGP LDS.
  const size_t dyn_lds = (size_t)NSEQ * 64 * sizeof(f16);
  msa_colglob_qkvo_kernel<<<NRES, 256, dyn_lds, stream>>>(
      m, mask, ln_w, ln_b, w_q, w_k, w_v, o_ws);
  msa_colglob_out_kernel<<<dim3(NSEQ / 128, NRES), 256, 0, stream>>>(
      m, ln_w, ln_b, w_g, b_g, w_o, b_o, o_ws, out);
}
